// DCTQuantify_4690104287843
// MI455X (gfx1250) — matrix operand order repaired and checked by reading, not compiled, not run on hardware
//
#include <hip/hip_runtime.h>

typedef __attribute__((ext_vector_type(16))) _Float16 v16h;
typedef __attribute__((ext_vector_type(8)))  float    v8f;
typedef __attribute__((ext_vector_type(4)))  float    v4f;

#define BB 32
#define CC 3
#define HH 512
#define WW 512

// f32 D-layout (8 regs) -> f16 A-layout of the TRANSPOSED matrix.
// Lane j holds D[0..7][j] (lanes<16) / D[8..15][j] (lanes>=16), which is
// exactly the A-layout for A = D^T with K padded 16->32 with zeros.
__device__ __forceinline__ v16h cvt_dT_to_a(v8f d) {
  v16h a = (v16h)0;
#pragma unroll
  for (int i = 0; i < 8; ++i) a[i] = (_Float16)d[i];
  return a;
}

__global__ __launch_bounds__(256) void dct_quant_kernel(
    const float* __restrict__ input,
    const float* __restrict__ dct_t,      // (8,8,8,8): C[x][u] = dct_t[x*512 + u*8]
    const float* __restrict__ dct_scale,  // (8,8)
    const float* __restrict__ idct_alpha, // (8,8)
    const float* __restrict__ y_table,    // (8,8) (transposed by setup)
    const float* __restrict__ c_table,    // (8,8)
    const int*   __restrict__ factor,     // scalar
    float* __restrict__ out)
{
  // Block-invariant constants in LDS (channel is uniform per block:
  // 1024 tiles per plane / 8 waves per block = 128 blocks per plane).
  __shared__ v16h sBc[32];   // B-layout columns of Chat  (K rows 16..31 zeroed)
  __shared__ v16h sBct[32];  // B-layout columns of Chat^T
  __shared__ __align__(16) float sPre[64];   // 127.5 * dct_scale / qt
  __shared__ __align__(16) float sPost[64];  // qt * idct_alpha * 0.25 / 127.5

  const int tid = threadIdx.x;
  const int ch  = (blockIdx.x >> 7) % CC;

  {
    _Float16* bc  = (_Float16*)sBc;
    _Float16* bct = (_Float16*)sBct;
#pragma unroll
    for (int e = tid; e < 512; e += 256) {
      const int l = e >> 4;   // lane 0..31
      const int k = ((l >> 4) << 3) + (e & 15);   // K = 8*(l/16) + s (table order), valid for s < 8
      float vc = 0.0f, vct = 0.0f;
      if ((e & 15) < 8 && !((k ^ l) & 8)) {           // blockdiag(C,C) structure
        vc  = dct_t[(k & 7) * 512 + (l & 7) * 8];     // Chat[k][l]
        vct = dct_t[(l & 7) * 512 + (k & 7) * 8];     // Chat^T[k][l]
      }
      bc[e]  = (_Float16)vc;
      bct[e] = (_Float16)vct;
    }
    if (tid < 64) {
      const float f  = (float)factor[0];
      const float fs = (ch == 0) ? f : 0.5f * f;
      const float qt = ((ch == 0) ? y_table[tid] : c_table[tid]) * fs;
      sPre[tid]  = 127.5f * dct_scale[tid] / qt;            // amortized over 8 waves
      sPost[tid] = qt * idct_alpha[tid] * (0.25f / 127.5f);
    }
  }
  __syncthreads();

  const int lane   = tid & 31;
  const int waveId = blockIdx.x * (blockDim.x >> 5) + (tid >> 5);

  const int tileIdx = waveId & 1023;     // 32x32 tiles of 16x16 per plane
  const int img     = waveId >> 10;      // b*3 + c
  const int ty = tileIdx >> 5, tx = tileIdx & 31;

  const int row     = lane & 15;         // tile-local h
  const int colBase = (lane >> 4) << 3;  // 0 or 8 (tile-local w)
  const size_t base = (size_t)img * (HH * WW) + (size_t)(ty * 16 + row) * WW
                    + (size_t)(tx * 16 + colBase);

  // ---- streaming load, A1 = x (16x16 f16, K padded to 32); 127.5 folded into pre ----
  const v4f x0 = __builtin_nontemporal_load((const v4f*)(input + base));
  const v4f x1 = __builtin_nontemporal_load((const v4f*)(input + base + 4));
  v16h a1 = (v16h)0;
#pragma unroll
  for (int i = 0; i < 4; ++i) a1[i]     = (_Float16)x0[i];
#pragma unroll
  for (int i = 0; i < 4; ++i) a1[4 + i] = (_Float16)x1[i];

  // ---- per-lane constants from LDS ----
  const v16h b_c  = sBc[lane];
  const v16h b_ct = sBct[lane];
  const int  u8   = lane & 7;            // s2[r] = D[u=lane&15][v=r(+8)] -> u8=lane&7, v8=r
  const v4f pre0  = *(const v4f*)&sPre[u8 * 8];
  const v4f pre1  = *(const v4f*)&sPre[u8 * 8 + 4];
  const v4f post0 = *(const v4f*)&sPost[u8 * 8];
  const v4f post1 = *(const v4f*)&sPost[u8 * 8 + 4];

  const v8f z = (v8f)0;
  // S1 = x * Chat
  v8f s1 = __builtin_amdgcn_wmma_f32_16x16x32_f16(false, a1, false, b_c, (short)0, z, false, false);
  // S2 = S1^T * Chat = Chat^T x^T Chat = D^T / 127.5  (free transpose via D->A relayout)
  v16h a2 = cvt_dT_to_a(s1);
  v8f s2 = __builtin_amdgcn_wmma_f32_16x16x32_f16(false, a2, false, b_c, (short)0, z, false, false);

  // ---- quantize / diff-round / dequantize (elementwise on D^T) ----
  v8f e;
#pragma unroll
  for (int r = 0; r < 8; ++r) {
    const float pr  = (r < 4) ? pre0[r & 3] : pre1[r & 3];
    const float po  = (r < 4) ? post0[r & 3] : post1[r & 3];
    const float qin = s2[r] * pr;
    const float rn  = rintf(qin);          // round-half-even, matches jnp.round
    const float d   = qin - rn;
    e[r] = (rn + d * d * d) * po;          // diff_round, dequant, alpha, 0.25/127.5 folded
  }

  // S3 = (dq*alpha) * Chat^T   (A3 = E^T, free transpose)
  v16h a3 = cvt_dT_to_a(e);
  v8f s3 = __builtin_amdgcn_wmma_f32_16x16x32_f16(false, a3, false, b_ct, (short)0, z, false, false);
  // S4 = S3^T * Chat^T = (Chat dq*alpha Chat^T)^T scaled to output units
  v16h a4 = cvt_dT_to_a(s3);
  v8f s4 = __builtin_amdgcn_wmma_f32_16x16x32_f16(false, a4, false, b_ct, (short)0, z, false, false);

  // ---- epilogue: clip to [-1,1] (offset/scale all pre-folded); streaming store ----
  v4f o0, o1;
#pragma unroll
  for (int r = 0; r < 4; ++r) o0[r] = fminf(fmaxf(s4[r],     -1.0f), 1.0f);
#pragma unroll
  for (int r = 0; r < 4; ++r) o1[r] = fminf(fmaxf(s4[4 + r], -1.0f), 1.0f);
  __builtin_nontemporal_store(o0, (v4f*)(out + base));
  __builtin_nontemporal_store(o1, (v4f*)(out + base + 4));
}

extern "C" void kernel_launch(void* const* d_in, const int* in_sizes, int n_in,
                              void* d_out, int out_size, void* d_ws, size_t ws_size,
                              hipStream_t stream) {
  (void)in_sizes; (void)n_in; (void)out_size; (void)d_ws; (void)ws_size;
  const float* input      = (const float*)d_in[0];
  const float* dct_tensor = (const float*)d_in[1];
  const float* dct_scale  = (const float*)d_in[2];
  // d_in[3] = idct_tensor: same cosine basis, derived from dct_tensor, unused.
  const float* idct_alpha = (const float*)d_in[4];
  const float* y_table    = (const float*)d_in[5];
  const float* c_table    = (const float*)d_in[6];
  const int*   factor     = (const int*)d_in[7];
  float* outp = (float*)d_out;

  // one wave per 16x16 tile: 32*3 planes * (512/16)^2 tiles = 98304 waves
  const int waves = BB * CC * (HH / 16) * (WW / 16);
  const int block = 256;                       // 8 waves per block
  const int grid  = waves / (block / 32);
  dct_quant_kernel<<<grid, block, 0, stream>>>(input, dct_tensor, dct_scale,
                                               idct_alpha, y_table, c_table,
                                               factor, outp);
}
